// StockForecastGNN_39599598469910
// MI455X (gfx1250) — matrix operand order repaired and checked by reading, not compiled, not run on hardware
//
#include <hip/hip_runtime.h>
#include <hip/hip_bf16.h>
#include <math.h>

#define N_NODES 10000
#define T_LEN   16
#define N_FEAT  32
#define HID     128
#define HEADS   4
#define EDGE_DIM 8
#define N_Q     3
#define N_EDGES 640000
#define E_TOT   (N_EDGES + N_NODES)
#define GATE    (4 * HID)   // 512

typedef __attribute__((ext_vector_type(16))) _Float16     v16h;
typedef __attribute__((ext_vector_type(8)))  float        v8f;
typedef __attribute__((ext_vector_type(4)))  unsigned int u32v4;

union HFrag { v16h v; u32v4 q[2]; };

// ---------------- WMMA fragment helpers (CDNA5 16x16x32 f16 layouts) ----------------

__device__ __forceinline__ v16h frag2(const void* p0, const void* p1) {
  HFrag u;
  u.q[0] = *(const u32v4*)p0;
  u.q[1] = *(const u32v4*)p1;
  return u.v;
}

// A fragment: row-major [16][strideHalf] f16, M = lane&15.
// lane<16: elems 0..7 -> K=k0..k0+7,  8..15 -> K=k0+16..k0+23
// lane>=16: elems 0..7 -> K=k0+8..+15, 8..15 -> K=k0+24..+31
__device__ __forceinline__ v16h load_a16(const _Float16* A, int strideHalf, int k0) {
  int lane = threadIdx.x & 31;
  const _Float16* r = A + (size_t)(lane & 15) * strideHalf + k0 + ((lane >> 4) << 3);
  return frag2(r, r + 16);
}

// B fragment: W row-major [Nout][ldb] f16; column N = n0+(lane&15) is row of W.
// lane<16: elems 0..7 -> K=k0..k0+7, 8..15 -> K=k0+16..k0+23 ; lane>=16: K=k0+8..+15, k0+24..+31
__device__ __forceinline__ v16h load_b16(const _Float16* W, int ldb, int n0, int k0) {
  int lane = threadIdx.x & 31;
  const _Float16* r = W + (size_t)(n0 + (lane & 15)) * ldb + k0 + ((lane >> 4) << 3);
  return frag2(r, r + 16);
}

__device__ __forceinline__ v8f wmma_f16(v16h a, v16h b, v8f c) {
  return __builtin_amdgcn_wmma_f32_16x16x32_f16(false, a, false, b, (short)0, c, false, false);
}

template <int K>
__device__ __forceinline__ v8f gemm16(const _Float16* A, int astr,
                                      const _Float16* W, int ldb, int n0, v8f acc) {
#pragma unroll
  for (int k0 = 0; k0 < K; k0 += 32)
    acc = wmma_f16(load_a16(A, astr, k0), load_b16(W, ldb, n0, k0), acc);
  return acc;
}

// ---------------- gfx1250 async global->LDS (ASYNCcnt-tracked) ----------------

__device__ __forceinline__ void async_ld16(unsigned lds_off, const void* gptr) {
  unsigned long long ga = (unsigned long long)gptr;
  asm volatile("global_load_async_to_lds_b128 %0, %1, off"
               :: "v"(lds_off), "v"(ga)
               : "memory");
}

__device__ __forceinline__ void wait_async0() {
  asm volatile("s_wait_asynccnt 0x0" ::: "memory");
}

__device__ __forceinline__ float sigf(float x) { return 1.0f / (1.0f + __expf(-x)); }

__device__ __forceinline__ void atomicMaxF(float* a, float v) {
  unsigned int* ai = (unsigned int*)a;
  unsigned int old = *ai;
  while (__uint_as_float(old) < v) {
    unsigned int prev = atomicCAS(ai, old, __float_as_uint(v));
    if (prev == old) break;
    old = prev;
  }
}

// ---------------- small prep kernels ----------------

__global__ void k_pack(const float* conv_w, const float* w_ih, const float* w_hh,
                       const float* b_ih, const float* b_hh, const float* gat_lin_w,
                       const float* ca_wv, const float* ca_wo, const float* tf_in_w,
                       const float* tf_out_w, const float* ff1_w, const float* ff2_w,
                       _Float16* convB, _Float16* wih16, _Float16* whh16, float* ball,
                       _Float16* gatw16, _Float16* cawv16, _Float16* cawo16,
                       _Float16* tfv16, _Float16* tfo16, _Float16* ff1w16, _Float16* ff2w16) {
  for (int i = blockIdx.x * blockDim.x + threadIdx.x; i < 65536;
       i += gridDim.x * blockDim.x) {
    if (i < 128 * 96) {
      int o = i / 96, k = i % 96, tap = k / 32, f = k % 32;
      convB[i] = (_Float16)conv_w[(o * 32 + f) * 3 + tap];   // [O][I][3] -> [O][tap*32+f]
    }
    wih16[i] = (_Float16)w_ih[i];
    whh16[i] = (_Float16)w_hh[i];
    if (i < GATE) ball[i] = b_ih[i] + b_hh[i];
    if (i < HID * HID) {
      gatw16[i] = (_Float16)gat_lin_w[i];
      cawv16[i] = (_Float16)ca_wv[i];
      cawo16[i] = (_Float16)ca_wo[i];
      tfv16[i]  = (_Float16)tf_in_w[2 * HID * HID + i];      // rows 256..383 (v slice)
      tfo16[i]  = (_Float16)tf_out_w[i];
    }
    if (i < 2 * HID * HID) {
      ff1w16[i] = (_Float16)ff1_w[i];
      ff2w16[i] = (_Float16)ff2_w[i];
    }
  }
}

__global__ void k_init(float* agg, float* amax, float* denom, float* cnt) {
  int i = blockIdx.x * blockDim.x + threadIdx.x;
  if (i < N_NODES * HID) agg[i] = 0.0f;
  if (i < N_NODES * HEADS) { amax[i] = -3.0e38f; denom[i] = 0.0f; }
  if (i < N_NODES) cnt[i] = 0.0f;
}

__global__ void k_mean(const float* ea, float* meanE) {
  __shared__ float red[256][EDGE_DIM];
  float acc[EDGE_DIM];
#pragma unroll
  for (int d = 0; d < EDGE_DIM; ++d) acc[d] = 0.0f;
  for (int e = threadIdx.x; e < N_EDGES; e += 256)
#pragma unroll
    for (int d = 0; d < EDGE_DIM; ++d) acc[d] += ea[(size_t)e * EDGE_DIM + d];
#pragma unroll
  for (int d = 0; d < EDGE_DIM; ++d) red[threadIdx.x][d] = acc[d];
  __syncthreads();
  if (threadIdx.x < EDGE_DIM) {
    float s = 0.0f;
    for (int j = 0; j < 256; ++j) s += red[j][threadIdx.x];
    meanE[threadIdx.x] = s / (float)N_EDGES;
  }
}

__global__ void k_weff(const float* Wedge, const float* att_edge, const float* meanE,
                       float* weff, float* loopE) {
  int t = threadIdx.x;
  if (t < HEADS * EDGE_DIM) {
    int h = t / EDGE_DIM, d = t % EDGE_DIM;
    float s = 0.0f;
    for (int c = 0; c < 32; ++c)
      s += att_edge[h * 32 + c] * Wedge[(h * 32 + c) * EDGE_DIM + d];
    weff[t] = s;
  }
  __syncthreads();
  if (t < HEADS) {
    float s = 0.0f;
    for (int d = 0; d < EDGE_DIM; ++d) s += meanE[d] * weff[t * EDGE_DIM + d];
    loopE[t] = s;
  }
}

// ---------------- Conv1d(k=3,pad=1)+ReLU via WMMA (im2col K=96) ----------------

__global__ void __launch_bounds__(256) k_conv(const float* x, const _Float16* WB,
                                              const float* conv_b, _Float16* hconv) {
  __shared__ _Float16 Aim[8][16 * 96];
  int tid = threadIdx.x, wave = tid >> 5, lane = tid & 31;
  int node = blockIdx.x * 8 + wave;
  _Float16* A = Aim[wave];
  for (int i = lane; i < 16 * 96; i += 32) {
    int t = i / 96, k = i % 96, tap = k / 32, f = k % 32, tt = t + tap - 1;
    float v = (tt >= 0 && tt < T_LEN) ? x[((size_t)node * T_LEN + tt) * N_FEAT + f] : 0.0f;
    A[i] = (_Float16)v;
  }
  __syncthreads();
  int lane15 = lane & 15, hi = lane >> 4;
  // hoist A fragments: reused by all 8 column tiles
  v16h af[3];
#pragma unroll
  for (int kc = 0; kc < 3; ++kc) af[kc] = load_a16(A, 96, kc * 32);
  for (int j = 0; j < 8; ++j) {
    int n0 = 16 * j;
    v8f acc = {0.f, 0.f, 0.f, 0.f, 0.f, 0.f, 0.f, 0.f};
#pragma unroll
    for (int kc = 0; kc < 3; ++kc)
      acc = wmma_f16(af[kc], load_b16(WB, 96, n0, kc * 32), acc);
    int n = n0 + lane15;
    float bn = conv_b[n];
#pragma unroll
    for (int r = 0; r < 8; ++r) {
      int m = r + 8 * hi;
      float v = acc[r] + bn;
      v = v > 0.0f ? v : 0.0f;
      hconv[((size_t)node * T_LEN + m) * HID + n] = (_Float16)v;
    }
  }
}

// ---- LSTM: weight-stationary in VGPRs, async double-buffered x-tiles ----
// Each wave owns 64 gate columns; its 32 B fragments (w_ih + w_hh, 4 col-tiles x
// 4 K-chunks) are loaded ONCE and held in registers across all 16 timesteps
// (CDNA5 waves address up to 1024 VGPRs). Inner loop = LDS A-loads + v_wmma only.

__global__ void __launch_bounds__(256) k_lstm(const _Float16* hconv, const _Float16* wih,
                                              const _Float16* whh, const float* ball,
                                              float* temporal32, _Float16* temporal16) {
  __shared__ _Float16 hst[16 * HID];
  __shared__ float gates[16 * GATE];
  __shared__ _Float16 xbuf[2][16 * HID];
  int tid = threadIdx.x, wave = tid >> 5, lane = tid & 31;
  int node0 = blockIdx.x * 16;
  int lane15 = lane & 15, hi = lane >> 4;
  float creg[8];
#pragma unroll
  for (int q = 0; q < 8; ++q) creg[q] = 0.0f;
  for (int i = tid; i < 16 * HID; i += 256) hst[i] = (_Float16)0.0f;

  int arow = tid >> 4, achk = tid & 15;  // 256 threads x 16B cover one 16x128 f16 tile
  // preload x tile for t=0 via async global->LDS
  async_ld16((unsigned)(uintptr_t)&xbuf[0][arow * HID + achk * 8],
             hconv + ((size_t)(node0 + arow) * T_LEN + 0) * HID + achk * 8);

  // weight-stationary B fragments (loop-invariant across the recurrence)
  v16h bih[4][4], bhh[4][4];
#pragma unroll
  for (int j = 0; j < 4; ++j)
#pragma unroll
    for (int kc = 0; kc < 4; ++kc) {
      bih[j][kc] = load_b16(wih, HID, wave * 64 + 16 * j, kc * 32);
      bhh[j][kc] = load_b16(whh, HID, wave * 64 + 16 * j, kc * 32);
    }

  wait_async0();
  __syncthreads();

  int cur = 0;
  for (int t = 0; t < T_LEN; ++t) {
    if (t + 1 < T_LEN)  // prefetch next step's x tile while computing this one
      async_ld16((unsigned)(uintptr_t)&xbuf[cur ^ 1][arow * HID + achk * 8],
                 hconv + ((size_t)(node0 + arow) * T_LEN + (t + 1)) * HID + achk * 8);
    // hoist A fragments (shared across the 4 column tiles of this wave)
    v16h ax[4], ah[4];
#pragma unroll
    for (int kc = 0; kc < 4; ++kc) {
      ax[kc] = load_a16(xbuf[cur], HID, kc * 32);
      ah[kc] = load_a16(hst, HID, kc * 32);
    }
#pragma unroll
    for (int j = 0; j < 4; ++j) {
      v8f acc = {0.f, 0.f, 0.f, 0.f, 0.f, 0.f, 0.f, 0.f};
#pragma unroll
      for (int kc = 0; kc < 4; ++kc) acc = wmma_f16(ax[kc], bih[j][kc], acc);
#pragma unroll
      for (int kc = 0; kc < 4; ++kc) acc = wmma_f16(ah[kc], bhh[j][kc], acc);
      int n = wave * 64 + 16 * j + lane15;
#pragma unroll
      for (int r = 0; r < 8; ++r) gates[(r + 8 * hi) * GATE + n] = acc[r];
    }
    __syncthreads();
#pragma unroll
    for (int q = 0; q < 8; ++q) {
      int idx = tid * 8 + q, m = idx >> 7, ch = idx & 127;
      float gi = gates[m * GATE + ch] + ball[ch];
      float gf = gates[m * GATE + HID + ch] + ball[HID + ch];
      float gg = gates[m * GATE + 2 * HID + ch] + ball[2 * HID + ch];
      float go = gates[m * GATE + 3 * HID + ch] + ball[3 * HID + ch];
      float c = sigf(gf) * creg[q] + sigf(gi) * tanhf(gg);
      creg[q] = c;
      float h = sigf(go) * tanhf(c);
      hst[m * HID + ch] = (_Float16)h;
      if (t == T_LEN - 1) {
        temporal32[(size_t)(node0 + m) * HID + ch] = h;
        temporal16[(size_t)(node0 + m) * HID + ch] = (_Float16)h;
      }
    }
    wait_async0();   // next x tile landed in LDS
    __syncthreads();
    cur ^= 1;
  }
}

// ---------------- GAT linear + per-head attention dots ----------------

__global__ void __launch_bounds__(256) k_gatlin(const _Float16* temporal16,
                                                const _Float16* gatw, const float* att_src,
                                                const float* att_dst, float* xs,
                                                float* asrc, float* adst) {
  __shared__ float xsl[16 * HID];
  int tid = threadIdx.x, wave = tid >> 5, lane = tid & 31;
  int node0 = blockIdx.x * 16;
  int n0 = wave * 16, lane15 = lane & 15, hi = lane >> 4;
  v8f acc = {0.f, 0.f, 0.f, 0.f, 0.f, 0.f, 0.f, 0.f};
  acc = gemm16<HID>(temporal16 + (size_t)node0 * HID, HID, gatw, HID, n0, acc);
  int n = n0 + lane15;
#pragma unroll
  for (int r = 0; r < 8; ++r) xsl[(r + 8 * hi) * HID + n] = acc[r];
  __syncthreads();
  for (int i = tid; i < 16 * HID; i += 256)
    xs[(size_t)(node0 + (i >> 7)) * HID + (i & 127)] = xsl[i];
  if (tid < 64) {
    int m = tid >> 2, h = tid & 3;
    float ss = 0.0f, sd = 0.0f;
    for (int c = 0; c < 32; ++c) {
      float v = xsl[m * HID + h * 32 + c];
      ss += v * att_src[h * 32 + c];
      sd += v * att_dst[h * 32 + c];
    }
    asrc[(size_t)(node0 + m) * HEADS + h] = ss;
    adst[(size_t)(node0 + m) * HEADS + h] = sd;
  }
}

// ---------------- edge softmax + aggregation ----------------

__global__ void k_alpha(const int* ei, const float* ea, const float* weff,
                        const float* loopE, const float* asrc, const float* adst,
                        float* alphaW, float* amax, float* cnt) {
  int e = blockIdx.x * blockDim.x + threadIdx.x;
  if (e >= E_TOT) return;
  int s, d;
  float ae[HEADS];
  if (e < N_EDGES) {
    s = ei[e];
    d = ei[N_EDGES + e];
    float a[EDGE_DIM];
#pragma unroll
    for (int dd = 0; dd < EDGE_DIM; ++dd) a[dd] = ea[(size_t)e * EDGE_DIM + dd];
#pragma unroll
    for (int h = 0; h < HEADS; ++h) {
      float t = 0.0f;
#pragma unroll
      for (int dd = 0; dd < EDGE_DIM; ++dd) t += a[dd] * weff[h * EDGE_DIM + dd];
      ae[h] = t;
    }
  } else {
    s = d = e - N_EDGES;
#pragma unroll
    for (int h = 0; h < HEADS; ++h) ae[h] = loopE[h];
  }
  atomicAdd(&cnt[d], 1.0f);
#pragma unroll
  for (int h = 0; h < HEADS; ++h) {
    float al = asrc[(size_t)s * HEADS + h] + adst[(size_t)d * HEADS + h] + ae[h];
    al = al > 0.0f ? al : 0.2f * al;  // leaky_relu 0.2
    alphaW[(size_t)e * HEADS + h] = al;
    atomicMaxF(&amax[(size_t)d * HEADS + h], al);
  }
}

__global__ void k_ex(const int* ei, float* alphaW, const float* amax, float* denom) {
  int e = blockIdx.x * blockDim.x + threadIdx.x;
  if (e >= E_TOT) return;
  int d = (e < N_EDGES) ? ei[N_EDGES + e] : e - N_EDGES;
#pragma unroll
  for (int h = 0; h < HEADS; ++h) {
    float ex = __expf(alphaW[(size_t)e * HEADS + h] - amax[(size_t)d * HEADS + h]);
    alphaW[(size_t)e * HEADS + h] = ex;
    atomicAdd(&denom[(size_t)d * HEADS + h], ex);
  }
}

__global__ void k_msg(const int* ei, const float* alphaW, const float* denom,
                      const float* xs, float* agg) {
  int e = blockIdx.x * blockDim.x + threadIdx.x;
  if (e >= E_TOT) return;
  int s = (e < N_EDGES) ? ei[e] : e - N_EDGES;
  int d = (e < N_EDGES) ? ei[N_EDGES + e] : e - N_EDGES;
  const float* xr = xs + (size_t)s * HID;
  __builtin_prefetch(xr, 0, 0);                 // global_prefetch_b8 for the gather row
  __builtin_prefetch(xr + 64, 0, 0);
  float attn[HEADS];
#pragma unroll
  for (int h = 0; h < HEADS; ++h)
    attn[h] = alphaW[(size_t)e * HEADS + h] / denom[(size_t)d * HEADS + h];
  float* ar = agg + (size_t)d * HID;
  for (int c = 0; c < HID; ++c) atomicAdd(&ar[c], xr[c] * attn[c >> 5]);
}

__global__ void k_graph(const float* agg, const float* cnt, const float* gat_bias,
                        _Float16* graph16) {
  int i = blockIdx.x * blockDim.x + threadIdx.x;
  if (i >= N_NODES * HID) return;
  float g = agg[i] / cnt[i >> 7] + gat_bias[i & 127];
  g = g > 0.0f ? g : (__expf(g) - 1.0f);  // ELU
  graph16[i] = (_Float16)g;
}

// ---------------- fused head: CA(v,wo) + TF(v2,out) + LN + FF + LN + proj ----------------

__global__ void __launch_bounds__(256) k_head(
    const _Float16* graph16, const float* temporal32, const _Float16* wv, const float* bv,
    const _Float16* wo, const float* bo, const _Float16* wv2, const float* bv2,
    const _Float16* wo2, const float* bo2, const float* ln1g, const float* ln1b,
    const _Float16* w1, const float* b1, const _Float16* w2, const float* b2,
    const float* ln2g, const float* ln2b, const float* outw, const float* outb,
    float* outp) {
  __shared__ _Float16 X16[16 * HID];
  __shared__ _Float16 Y16[16 * HID];
  __shared__ _Float16 Z16[16 * 2 * HID];
  __shared__ float FU[16 * HID];
  __shared__ float YY[16 * HID];
  __shared__ float SC[16 * HID];
  __shared__ float mu[16], iv[16];
  int tid = threadIdx.x, wave = tid >> 5, lane = tid & 31;
  int node0 = blockIdx.x * 16;
  int n0 = wave * 16, lane15 = lane & 15, hi = lane >> 4;
  int nl = n0 + lane15;
  const v8f Z = {0.f, 0.f, 0.f, 0.f, 0.f, 0.f, 0.f, 0.f};
  v8f acc;

  // s1: v = graph @ wv^T + bv  (cross-attn softmax over len-1 == 1, q/k dead)
  acc = gemm16<HID>(graph16 + (size_t)node0 * HID, HID, wv, HID, n0, Z);
  {
    float b = bv[nl];
#pragma unroll
    for (int r = 0; r < 8; ++r) X16[(r + 8 * hi) * HID + nl] = (_Float16)(acc[r] + b);
  }
  __syncthreads();
  // s2: fused = temporal + v @ wo^T + bo
  acc = gemm16<HID>(X16, HID, wo, HID, n0, Z);
  {
    float b = bo[nl];
#pragma unroll
    for (int r = 0; r < 8; ++r) {
      int m = r + 8 * hi;
      float v = temporal32[(size_t)(node0 + m) * HID + nl] + acc[r] + b;
      FU[m * HID + nl] = v;
      Y16[m * HID + nl] = (_Float16)v;
    }
  }
  __syncthreads();
  // s3: v2 = fused @ tf_in_w[2H:3H]^T + b  (self-attn softmax over len-1 == 1)
  acc = gemm16<HID>(Y16, HID, wv2, HID, n0, Z);
  {
    float b = bv2[nl];
#pragma unroll
    for (int r = 0; r < 8; ++r) X16[(r + 8 * hi) * HID + nl] = (_Float16)(acc[r] + b);
  }
  __syncthreads();
  // s4: sa = v2 @ tf_out_w^T + b ; SC = fused + sa
  acc = gemm16<HID>(X16, HID, wo2, HID, n0, Z);
  {
    float b = bo2[nl];
#pragma unroll
    for (int r = 0; r < 8; ++r) {
      int m = r + 8 * hi;
      SC[m * HID + nl] = FU[m * HID + nl] + acc[r] + b;
    }
  }
  __syncthreads();
  // LN1
  if (tid < 16) {
    float s = 0.0f;
    for (int c = 0; c < HID; ++c) s += SC[tid * HID + c];
    float m_ = s / (float)HID, v = 0.0f;
    for (int c = 0; c < HID; ++c) { float dl = SC[tid * HID + c] - m_; v += dl * dl; }
    mu[tid] = m_;
    iv[tid] = rsqrtf(v / (float)HID + 1e-5f);
  }
  __syncthreads();
  for (int i = tid; i < 16 * HID; i += 256) {
    int m = i >> 7, ch = i & 127;
    float y = (SC[i] - mu[m]) * iv[m] * ln1g[ch] + ln1b[ch];
    YY[i] = y;
    Y16[i] = (_Float16)y;
  }
  __syncthreads();
  // s5: ff1 = relu(y @ ff1_w^T + b1)  [16 x 256]  (A fragments hoisted over halves)
  {
    v16h ay[4];
#pragma unroll
    for (int kc = 0; kc < 4; ++kc) ay[kc] = load_a16(Y16, HID, kc * 32);
    for (int half = 0; half < 2; ++half) {
      int c0 = n0 + half * HID;
      acc = Z;
#pragma unroll
      for (int kc = 0; kc < 4; ++kc)
        acc = wmma_f16(ay[kc], load_b16(w1, HID, c0, kc * 32), acc);
      int nc = c0 + lane15;
      float b = b1[nc];
#pragma unroll
      for (int r = 0; r < 8; ++r) {
        int m = r + 8 * hi;
        float v = acc[r] + b;
        v = v > 0.0f ? v : 0.0f;
        Z16[m * 2 * HID + nc] = (_Float16)v;
      }
    }
  }
  __syncthreads();
  // s6: ff = ff1 @ ff2_w^T + b2 (K=256) ; SC = y + ff
  acc = gemm16<2 * HID>(Z16, 2 * HID, w2, 2 * HID, n0, Z);
  {
    float b = b2[nl];
#pragma unroll
    for (int r = 0; r < 8; ++r) {
      int m = r + 8 * hi;
      SC[m * HID + nl] = YY[m * HID + nl] + acc[r] + b;
    }
  }
  __syncthreads();
  // LN2
  if (tid < 16) {
    float s = 0.0f;
    for (int c = 0; c < HID; ++c) s += SC[tid * HID + c];
    float m_ = s / (float)HID, v = 0.0f;
    for (int c = 0; c < HID; ++c) { float dl = SC[tid * HID + c] - m_; v += dl * dl; }
    mu[tid] = m_;
    iv[tid] = rsqrtf(v / (float)HID + 1e-5f);
  }
  __syncthreads();
  for (int i = tid; i < 16 * HID; i += 256) {
    int m = i >> 7, ch = i & 127;
    YY[i] = (SC[i] - mu[m]) * iv[m] * ln2g[ch] + ln2b[ch];
  }
  __syncthreads();
  // out projection [16 x 3]
  if (tid < 16 * N_Q) {
    int m = tid / N_Q, q = tid % N_Q;
    float s = outb[q];
    for (int c = 0; c < HID; ++c) s += YY[m * HID + c] * outw[q * HID + c];
    outp[(size_t)(node0 + m) * N_Q + q] = s;
  }
}

// ---------------- host launch ----------------

extern "C" void kernel_launch(void* const* d_in, const int* in_sizes, int n_in,
                              void* d_out, int out_size, void* d_ws, size_t ws_size,
                              hipStream_t stream) {
  (void)in_sizes; (void)n_in; (void)out_size; (void)ws_size;
  const float* x          = (const float*)d_in[0];
  const int*   ei         = (const int*)d_in[1];
  const float* ea         = (const float*)d_in[2];
  const float* conv_w     = (const float*)d_in[3];
  const float* conv_b     = (const float*)d_in[4];
  const float* w_ih       = (const float*)d_in[5];
  const float* w_hh       = (const float*)d_in[6];
  const float* b_ih       = (const float*)d_in[7];
  const float* b_hh       = (const float*)d_in[8];
  const float* gat_lin_w  = (const float*)d_in[9];
  const float* att_src    = (const float*)d_in[10];
  const float* att_dst    = (const float*)d_in[11];
  const float* gat_edge_w = (const float*)d_in[12];
  const float* att_edge   = (const float*)d_in[13];
  const float* gat_bias   = (const float*)d_in[14];
  const float* ca_wv      = (const float*)d_in[19];
  const float* ca_bv      = (const float*)d_in[20];
  const float* ca_wo      = (const float*)d_in[21];
  const float* ca_bo      = (const float*)d_in[22];
  const float* tf_in_w    = (const float*)d_in[23];
  const float* tf_in_b    = (const float*)d_in[24];
  const float* tf_out_w   = (const float*)d_in[25];
  const float* tf_out_b   = (const float*)d_in[26];
  const float* ln1_g      = (const float*)d_in[27];
  const float* ln1_b      = (const float*)d_in[28];
  const float* ff1_w      = (const float*)d_in[29];
  const float* ff1_b      = (const float*)d_in[30];
  const float* ff2_w      = (const float*)d_in[31];
  const float* ff2_b      = (const float*)d_in[32];
  const float* ln2_g      = (const float*)d_in[33];
  const float* ln2_b      = (const float*)d_in[34];
  const float* out_w      = (const float*)d_in[35];
  const float* out_b      = (const float*)d_in[36];

  char* ws = (char*)d_ws;
  size_t cur = 0;
  auto alloc = [&](size_t bytes) -> char* {
    char* p = ws + cur;
    cur = (cur + bytes + 255) & ~(size_t)255;
    return p;
  };

  _Float16* convB   = (_Float16*)alloc(128 * 96 * 2);
  _Float16* wih16   = (_Float16*)alloc((size_t)GATE * HID * 2);
  _Float16* whh16   = (_Float16*)alloc((size_t)GATE * HID * 2);
  float*    ball    = (float*)alloc(GATE * 4);
  _Float16* gatw16  = (_Float16*)alloc(HID * HID * 2);
  _Float16* cawv16  = (_Float16*)alloc(HID * HID * 2);
  _Float16* cawo16  = (_Float16*)alloc(HID * HID * 2);
  _Float16* tfv16   = (_Float16*)alloc(HID * HID * 2);
  _Float16* tfo16   = (_Float16*)alloc(HID * HID * 2);
  _Float16* ff1w16  = (_Float16*)alloc(2 * HID * HID * 2);
  _Float16* ff2w16  = (_Float16*)alloc(2 * HID * HID * 2);
  _Float16* hconv16 = (_Float16*)alloc((size_t)N_NODES * T_LEN * HID * 2);
  float*    temp32  = (float*)alloc((size_t)N_NODES * HID * 4);
  _Float16* temp16  = (_Float16*)alloc((size_t)N_NODES * HID * 2);
  float*    xs      = (float*)alloc((size_t)N_NODES * HID * 4);
  float*    asrc    = (float*)alloc((size_t)N_NODES * HEADS * 4);
  float*    adst    = (float*)alloc((size_t)N_NODES * HEADS * 4);
  float*    weff    = (float*)alloc(HEADS * EDGE_DIM * 4);
  float*    meanE   = (float*)alloc(EDGE_DIM * 4);
  float*    loopE   = (float*)alloc(HEADS * 4);
  float*    alphaW  = (float*)alloc((size_t)E_TOT * HEADS * 4);
  float*    amax    = (float*)alloc((size_t)N_NODES * HEADS * 4);
  float*    denom   = (float*)alloc((size_t)N_NODES * HEADS * 4);
  float*    cnt     = (float*)alloc((size_t)N_NODES * 4);
  float*    agg     = (float*)alloc((size_t)N_NODES * HID * 4);
  _Float16* graph16 = (_Float16*)alloc((size_t)N_NODES * HID * 2);

  const int TPB = 256;
  int tilesN = N_NODES / 16;                 // 625
  int gridE  = (E_TOT + TPB - 1) / TPB;
  int gridNH = (N_NODES * HID + TPB - 1) / TPB;

  k_pack<<<256, TPB, 0, stream>>>(conv_w, w_ih, w_hh, b_ih, b_hh, gat_lin_w, ca_wv, ca_wo,
                                  tf_in_w, tf_out_w, ff1_w, ff2_w, convB, wih16, whh16,
                                  ball, gatw16, cawv16, cawo16, tfv16, tfo16, ff1w16,
                                  ff2w16);
  k_init<<<gridNH, TPB, 0, stream>>>(agg, amax, denom, cnt);
  k_mean<<<1, TPB, 0, stream>>>(ea, meanE);
  k_weff<<<1, 64, 0, stream>>>(gat_edge_w, att_edge, meanE, weff, loopE);
  k_conv<<<N_NODES / 8, TPB, 0, stream>>>(x, convB, conv_b, hconv16);
  k_lstm<<<tilesN, TPB, 0, stream>>>(hconv16, wih16, whh16, ball, temp32, temp16);
  k_gatlin<<<tilesN, TPB, 0, stream>>>(temp16, gatw16, att_src, att_dst, xs, asrc, adst);
  k_alpha<<<gridE, TPB, 0, stream>>>(ei, ea, weff, loopE, asrc, adst, alphaW, amax, cnt);
  k_ex<<<gridE, TPB, 0, stream>>>(ei, alphaW, amax, denom);
  k_msg<<<gridE, TPB, 0, stream>>>(ei, alphaW, denom, xs, agg);
  k_graph<<<gridNH, TPB, 0, stream>>>(agg, cnt, gat_bias, graph16);
  k_head<<<tilesN, TPB, 0, stream>>>(graph16, temp32, cawv16, ca_bv, cawo16, ca_bo, tfv16,
                                     tf_in_b + 2 * HID, tfo16, tf_out_b, ln1_g, ln1_b,
                                     ff1w16, ff1_b, ff2w16, ff2_b, ln2_g, ln2_b, out_w,
                                     out_b, (float*)d_out);
}
